// WindowAttention_18064632447448
// MI455X (gfx1250) — matrix operand order repaired and checked by reading, not compiled, not run on hardware
//
#include <hip/hip_runtime.h>
#include <hip/hip_bf16.h>

// ---------------- constants ----------------
#define B_BATCH 2048
#define NTOK    49
#define DIMC    384
#define HEADS   12
#define HD      32
#define MROWS   (B_BATCH * NTOK)   // 100352 (exactly 6272 tiles of 16)
#define QKVN    (5 * DIMC)         // 1920
#define SCALE_F 0.17677669529663687f  // 32^-0.5

#define QKV_NT  5   // 16x80 strip per wave: 24 waves cover 120 n-tiles
#define PROJ_NT 6   // 16x96 strip per wave: 4 waves cover all 24 n-tiles

typedef __bf16 bf16;
typedef __bf16 v16bf __attribute__((ext_vector_type(16)));
typedef float  v8f   __attribute__((ext_vector_type(8)));

union AFrag { v16bf v; uint4 q[2]; };

__device__ __forceinline__ v8f wmma_bf16(v16bf a, v16bf b, v8f c) {
  // D = A(16x32 bf16) x B(32x16 bf16) + C(16x16 f32)
  return __builtin_amdgcn_wmma_f32_16x16x32_bf16(false, a, false, b, (short)0, c,
                                                 false, false);
}

// ---------------- kernel 1: weight f32 -> bf16 ----------------
__global__ void cvt_weights(const float* __restrict__ qkv_w,
                            const float* __restrict__ proj_w,
                            bf16* __restrict__ wq, bf16* __restrict__ wp) {
  int i = blockIdx.x * blockDim.x + threadIdx.x;
  if (i < QKVN * DIMC) wq[i] = (bf16)qkv_w[i];
  if (i < DIMC * DIMC) wp[i] = (bf16)proj_w[i];
}

// ---------------- kernel 2: QKV projection ----------------
// grid (6 ngroups, MROWS/16, 2 sources), block 128 (4 waves).
// Wave computes a 16x(16*QKV_NT) strip: A-frag converted once per K-step,
// reused across QKV_NT B-frags/WMMAs. Blocks sharing an mtile are launch-
// adjacent so the f32 A tile is HBM-fetched once and L2-hit afterwards.
__global__ void qkv_gemm(const float* __restrict__ x, const float* __restrict__ d,
                         const bf16* __restrict__ wq, const float* __restrict__ qkv_b,
                         bf16* __restrict__ qkvbuf) {
  const int lane = threadIdx.x & 31;
  const int wave = threadIdx.x >> 5;
  const int half = lane >> 4;
  const int l15  = lane & 15;
  const int mtile = blockIdx.y;
  const int nt0   = (blockIdx.x * 4 + wave) * QKV_NT;  // first n-tile (0..115)
  const int src   = blockIdx.z;
  const float* A  = src ? d : x;

  const int m = mtile * 16 + l15;            // A row owned by this lane

  v8f acc[QKV_NT];
#pragma unroll
  for (int j = 0; j < QKV_NT; ++j) acc[j] = (v8f){};

#pragma unroll
  for (int kb = 0; kb < DIMC / 32; ++kb) {
    // A fragment: rows in f32, convert to bf16 once, reuse QKV_NT times
    const float* ap = A + (size_t)m * DIMC + kb * 32 + half * 16;
    union { float4 f[4]; float s[16]; } t;
    t.f[0] = *(const float4*)(ap + 0);
    t.f[1] = *(const float4*)(ap + 4);
    t.f[2] = *(const float4*)(ap + 8);
    t.f[3] = *(const float4*)(ap + 12);
    AFrag a;
#pragma unroll
    for (int e = 0; e < 16; ++e) ((bf16*)&a.v)[e] = (bf16)t.s[e];

    if (kb + 1 < DIMC / 32)
      __builtin_prefetch(wq + (size_t)(nt0 * 16 + l15) * DIMC + (kb + 1) * 32, 0, 3);

#pragma unroll
    for (int j = 0; j < QKV_NT; ++j) {
      const int n = (nt0 + j) * 16 + l15;
      v16bf b = *(const v16bf*)(wq + (size_t)n * DIMC + kb * 32 + half * 16);
      acc[j] = wmma_bf16(a.v, b, acc[j]);
    }
  }

  // epilogue: scatter into [src,mat,B,H,N,hd] bf16
#pragma unroll
  for (int j = 0; j < QKV_NT; ++j) {
    const int n   = (nt0 + j) * 16 + l15;    // C/D column owned by this lane
    const float bias = qkv_b[n];
    const int mat = n / DIMC;
    const int h   = (n % DIMC) / HD;
    const int kd  = n % HD;
    const size_t matbase =
        (size_t)(src * 5 + mat) * B_BATCH * HEADS * (NTOK * HD);
#pragma unroll
    for (int r = 0; r < 8; ++r) {
      int mrow = mtile * 16 + r + half * 8;
      int bb = mrow / NTOK, tt = mrow % NTOK;
      size_t idx =
          matbase + ((size_t)bb * HEADS + h) * (NTOK * HD) + tt * HD + kd;
      qkvbuf[idx] = (bf16)(acc[j][r] + bias);
    }
  }
}

// ---------------- kernel 3: fused attention ----------------
// grid (B, HEADS, 2 groups), block 128 (4 waves, 16 S-rows each).
// group g: K/V from source g (mats 1,2); query sets:
//   s0: (src=g,   mat0) [+rel bias if g==0]
//   s1: (src=1-g, mat3)
//   s2: (src=g,   mat4)
// out: Aout[g] = s0+s1 result, Aout[2+g] = s2 result   ([B,N,H*hd] bf16)
__global__ void attn_kernel(const bf16* __restrict__ qkvbuf,
                            const float* __restrict__ rel_table,
                            const int* __restrict__ rel_index,
                            bf16* __restrict__ aout) {
  __shared__ __align__(32) bf16 Kb[64][32];      // K row-major, padded
  __shared__ __align__(32) bf16 Vt[32][64];      // V transposed  [hd][token]
  __shared__ __align__(32) bf16 Qb[3][64][32];   // 3 query sets
  __shared__ __align__(32) bf16 Pb[4][16][64];   // per-wave P tiles

  const int b = blockIdx.x, h = blockIdx.y, g = blockIdx.z;
  const int lane = threadIdx.x & 31;
  const int wave = threadIdx.x >> 5;
  const int half = lane >> 4;
  const int l15  = lane & 15;

  const size_t bh = (size_t)b * HEADS + h;
  const size_t matstride = (size_t)B_BATCH * HEADS * (NTOK * HD);
  const bf16* Kg = qkvbuf + (size_t)(g * 5 + 1) * matstride + bh * (NTOK * HD);
  const bf16* Vg = qkvbuf + (size_t)(g * 5 + 2) * matstride + bh * (NTOK * HD);
  const int qsrc[3] = {g, 1 - g, g};
  const int qmat[3] = {0, 3, 4};

  for (int idx = threadIdx.x; idx < 64 * 32; idx += 128) {
    int t = idx >> 5, kd = idx & 31;
    Kb[t][kd] = (t < NTOK) ? Kg[t * HD + kd] : (bf16)0.0f;
    int t2 = idx & 63, kd2 = idx >> 6;
    Vt[kd2][t2] = (t2 < NTOK) ? Vg[t2 * HD + kd2] : (bf16)0.0f;
#pragma unroll
    for (int s = 0; s < 3; ++s) {
      const bf16* Qg = qkvbuf + (size_t)(qsrc[s] * 5 + qmat[s]) * matstride +
                       bh * (NTOK * HD);
      Qb[s][t][kd] = (t < NTOK) ? Qg[t * HD + kd] : (bf16)0.0f;
    }
  }
  __syncthreads();

  v8f osum0 = {}, osum1 = {};

  for (int s = 0; s < 3; ++s) {
    // ---- gemm1: S = Q K^T  (one WMMA per 16x16 S tile, K-dim = 32) ----
    const int arow = wave * 16 + l15;
    AFrag aq;
    const bf16* qp = &Qb[s][arow][half * 16];
    aq.q[0] = *(const uint4*)(qp);
    aq.q[1] = *(const uint4*)(qp + 8);

    v8f sacc[4];
#pragma unroll
    for (int nt = 0; nt < 4; ++nt) {
      v16bf bk = *(const v16bf*)(&Kb[nt * 16 + l15][half * 16]);
      v8f z = {};
      sacc[nt] = wmma_bf16(aq.v, bk, z);
    }

    // ---- scale, relative-position bias, column mask ----
#pragma unroll
    for (int nt = 0; nt < 4; ++nt) {
      const int n = nt * 16 + l15;
      if (n < NTOK) {
#pragma unroll
        for (int r = 0; r < 8; ++r) {
          float v = sacc[nt][r] * SCALE_F;
          if (g == 0 && s == 0) {
            int mrow = wave * 16 + r + half * 8;
            if (mrow < NTOK)
              v += rel_table[(size_t)rel_index[mrow * NTOK + n] * HEADS + h];
          }
          sacc[nt][r] = v;
        }
      } else {
#pragma unroll
        for (int r = 0; r < 8; ++r) sacc[nt][r] = -1.0e30f;
      }
    }

    // ---- row softmax (rows live across 16 lanes of a half-wave) ----
    float rrec[8];
#pragma unroll
    for (int r = 0; r < 8; ++r) {
      float m = fmaxf(fmaxf(sacc[0][r], sacc[1][r]),
                      fmaxf(sacc[2][r], sacc[3][r]));
      m = fmaxf(m, __shfl_xor(m, 1, 32));
      m = fmaxf(m, __shfl_xor(m, 2, 32));
      m = fmaxf(m, __shfl_xor(m, 4, 32));
      m = fmaxf(m, __shfl_xor(m, 8, 32));
#pragma unroll
      for (int nt = 0; nt < 4; ++nt) sacc[nt][r] = __expf(sacc[nt][r] - m);
      float sum = sacc[0][r] + sacc[1][r] + sacc[2][r] + sacc[3][r];
      sum += __shfl_xor(sum, 1, 32);
      sum += __shfl_xor(sum, 2, 32);
      sum += __shfl_xor(sum, 4, 32);
      sum += __shfl_xor(sum, 8, 32);
      rrec[r] = 1.0f / sum;
    }

    // ---- route P through LDS to re-layout acc -> A fragment ----
    __syncthreads();  // protect Pb WAR vs previous set's reads
#pragma unroll
    for (int nt = 0; nt < 4; ++nt)
#pragma unroll
      for (int r = 0; r < 8; ++r)
        Pb[wave][r + half * 8][nt * 16 + l15] = (bf16)sacc[nt][r];
    __syncthreads();

    // ---- gemm2: O = P V  (K-dim 64 tokens = 2 WMMA steps) ----
    v8f oacc[2] = {{}, {}};
#pragma unroll
    for (int kb = 0; kb < 2; ++kb) {
      AFrag ap;
      const bf16* pp = &Pb[wave][l15][kb * 32 + half * 16];
      ap.q[0] = *(const uint4*)(pp);
      ap.q[1] = *(const uint4*)(pp + 8);
#pragma unroll
      for (int nt = 0; nt < 2; ++nt) {
        v16bf bv = *(const v16bf*)(&Vt[nt * 16 + l15][kb * 32 + half * 16]);
        oacc[nt] = wmma_bf16(ap.v, bv, oacc[nt]);
      }
    }
#pragma unroll
    for (int nt = 0; nt < 2; ++nt)
#pragma unroll
      for (int r = 0; r < 8; ++r) oacc[nt][r] *= rrec[r];

    if (s < 2) {
      osum0 += oacc[0];
      osum1 += oacc[1];
    } else {
      bf16* dst = aout + (size_t)(2 + g) * MROWS * DIMC;
#pragma unroll
      for (int nt = 0; nt < 2; ++nt)
#pragma unroll
        for (int r = 0; r < 8; ++r) {
          int mrow = wave * 16 + r + half * 8;
          if (mrow < NTOK)
            dst[((size_t)b * NTOK + mrow) * DIMC + h * HD + nt * 16 + l15] =
                (bf16)oacc[nt][r];
        }
    }
  }

  // write fused sum (sets 0+1) -> Aout[g]
  {
    bf16* dst = aout + (size_t)g * MROWS * DIMC;
    v8f os[2] = {osum0, osum1};
#pragma unroll
    for (int nt = 0; nt < 2; ++nt)
#pragma unroll
      for (int r = 0; r < 8; ++r) {
        int mrow = wave * 16 + r + half * 8;
        if (mrow < NTOK)
          dst[((size_t)b * NTOK + mrow) * DIMC + h * HD + nt * 16 + l15] =
              (bf16)os[nt][r];
      }
  }
}

// ---------------- kernel 4: output projection ----------------
// grid (MROWS/16, 4 outputs), block 128; each wave computes a 16x96 strip
// (PROJ_NT accumulators), so one block covers all 24 n-tiles and the bf16
// A row is fetched once per wave per K-step.
__global__ void proj_gemm(const bf16* __restrict__ aout,
                          const bf16* __restrict__ wp,
                          const float* __restrict__ proj_b,
                          float* __restrict__ out) {
  const int lane = threadIdx.x & 31;
  const int wave = threadIdx.x >> 5;
  const int half = lane >> 4;
  const int l15  = lane & 15;
  const int mtile = blockIdx.x;
  const int o     = blockIdx.y;
  const int nt0   = wave * PROJ_NT;  // 0,6,12,18

  const bf16* A = aout + (size_t)o * MROWS * DIMC;
  const int m = mtile * 16 + l15;
  const bf16* arow = A + (size_t)m * DIMC;

  v8f acc[PROJ_NT];
#pragma unroll
  for (int j = 0; j < PROJ_NT; ++j) acc[j] = (v8f){};

#pragma unroll
  for (int kb = 0; kb < DIMC / 32; ++kb) {
    AFrag a;
    const bf16* ap = arow + kb * 32 + half * 16;
    a.q[0] = *(const uint4*)(ap);
    a.q[1] = *(const uint4*)(ap + 8);

    if (kb + 1 < DIMC / 32)
      __builtin_prefetch(wp + (size_t)(nt0 * 16 + l15) * DIMC + (kb + 1) * 32, 0, 3);

#pragma unroll
    for (int j = 0; j < PROJ_NT; ++j) {
      const int n = (nt0 + j) * 16 + l15;
      v16bf b = *(const v16bf*)(wp + (size_t)n * DIMC + kb * 32 + half * 16);
      acc[j] = wmma_bf16(a.v, b, acc[j]);
    }
  }

  float* obase = out + (size_t)o * MROWS * DIMC;
#pragma unroll
  for (int j = 0; j < PROJ_NT; ++j) {
    const int n = (nt0 + j) * 16 + l15;
    const float pb = proj_b[n];
#pragma unroll
    for (int r = 0; r < 8; ++r) {
      int mrow = mtile * 16 + r + half * 8;
      obase[(size_t)mrow * DIMC + n] = acc[j][r] + pb;
    }
  }
}

// ---------------- launcher ----------------
extern "C" void kernel_launch(void* const* d_in, const int* in_sizes, int n_in,
                              void* d_out, int out_size, void* d_ws, size_t ws_size,
                              hipStream_t stream) {
  const float* x         = (const float*)d_in[0];
  const float* d         = (const float*)d_in[1];
  const float* qkv_w     = (const float*)d_in[2];
  const float* qkv_b     = (const float*)d_in[3];
  const float* rel_table = (const float*)d_in[4];
  const float* proj_w    = (const float*)d_in[5];
  const float* proj_b    = (const float*)d_in[6];
  const int*   rel_index = (const int*)d_in[7];
  float* out = (float*)d_out;

  // workspace carve-up (bf16, all 32B-aligned offsets)
  char* ws = (char*)d_ws;
  bf16* wq = (bf16*)ws;                                  // 1920*384
  ws += (size_t)QKVN * DIMC * sizeof(bf16);
  bf16* wp = (bf16*)ws;                                  // 384*384
  ws += (size_t)DIMC * DIMC * sizeof(bf16);
  bf16* qkvbuf = (bf16*)ws;                              // 10 * B*H*N*hd
  ws += (size_t)10 * B_BATCH * HEADS * NTOK * HD * sizeof(bf16);
  bf16* aout = (bf16*)ws;                                // 4 * MROWS*DIM
  ws += (size_t)4 * MROWS * DIMC * sizeof(bf16);

  // 1. weights -> bf16
  {
    int n = QKVN * DIMC;
    cvt_weights<<<(n + 255) / 256, 256, 0, stream>>>(qkv_w, proj_w, wq, wp);
  }
  // 2. QKV projections for both sources (L2-friendly: mtile in grid.y)
  {
    dim3 grid((QKVN / 16) / (4 * QKV_NT), MROWS / 16, 2);  // (6, 6272, 2)
    qkv_gemm<<<grid, 128, 0, stream>>>(x, d, wq, qkv_b, qkvbuf);
  }
  // 3. fused attention (6 attentions as 2 groups x 3 query sets)
  {
    dim3 grid(B_BATCH, HEADS, 2);
    attn_kernel<<<grid, 128, 0, stream>>>(qkvbuf, rel_table, rel_index, aout);
  }
  // 4. output projections
  {
    dim3 grid(MROWS / 16, 4);
    proj_gemm<<<grid, 128, 0, stream>>>(aout, wp, proj_b, out);
  }
}
